// Compress_88235808129265
// MI455X (gfx1250) — matrix operand order repaired and checked by reading, not compiled, not run on hardware
//
#include <hip/hip_runtime.h>

// ---------------------------------------------------------------------------
// NSA-style KV compression for MI455X (gfx1250, wave32, WMMA).
//   gates = sigmoid( (Gw[32,4096] . flat[w,h,4096]) * 4096^-0.5 )
//   out[w,h,d] = sum_k gates[w,h,k] * toks[w,k,h,d]
// Tile: 16 windows x 1 head per block; gate GEMM = 16x32x4096 bf16 WMMA.
// gate_weight is pre-converted to bf16 in d_ws so the hot loop is pure
// b128 loads + v_wmma.
// ---------------------------------------------------------------------------

typedef __attribute__((ext_vector_type(16))) __bf16 v16bf;
typedef __attribute__((ext_vector_type(8)))  float  v8f;

#define KWIN      32
#define HEAD_DIM  128
#define NUM_HEADS 2
#define FEAT      (KWIN * HEAD_DIM)   // 4096 features per (window, head)
#define MTILE     16                  // windows per block
#define RSTRIDE   (FEAT + 8)          // ushort row stride: +16B pad, keeps 16B align
#define NTHREADS  128                 // 4 wave32

// LDS carve-up (dynamic):
//   Abf   : MTILE * RSTRIDE ushorts (bf16 staged tile)     = 131,328 B
//   cpart : 4 waves * 512 floats (partial C matrices)      =   8,192 B
//   gl    : 16 * 32 floats (sigmoid gates)                 =   2,048 B
#define LDS_ABF_BYTES   (MTILE * RSTRIDE * 2)
#define LDS_CPART_FLTS  (4 * 512)
#define LDS_TOTAL_BYTES (LDS_ABF_BYTES + (LDS_CPART_FLTS + MTILE * KWIN) * 4)

__device__ __forceinline__ unsigned pack2_bf16(float a, float b) {
  unsigned ua = __float_as_uint(a);
  ua += 0x7FFFu + ((ua >> 16) & 1u);          // round-to-nearest-even
  unsigned ub = __float_as_uint(b);
  ub += 0x7FFFu + ((ub >> 16) & 1u);
  return (ua >> 16) | (ub & 0xFFFF0000u);
}

__device__ __forceinline__ float bf16_to_f32(unsigned short h) {
  return __uint_as_float(((unsigned)h) << 16);
}

// Initialize d_out from the (zero) compressed_buffer input.
__global__ void __launch_bounds__(NTHREADS)
compress_copy_init(const float* __restrict__ src, float* __restrict__ dst, long n) {
  long i = (long)blockIdx.x * blockDim.x + threadIdx.x;
  long stride = (long)gridDim.x * blockDim.x;
  for (; i < n; i += stride) dst[i] = src[i];
}

// One-time f32 -> packed bf16 conversion of gate_weight into d_ws.
__global__ void __launch_bounds__(NTHREADS)
gatew_to_bf16(const float* __restrict__ src, unsigned* __restrict__ dst, int npairs) {
  int i = blockIdx.x * blockDim.x + threadIdx.x;
  int stride = gridDim.x * blockDim.x;
  for (; i < npairs; i += stride) {
    const float2 v = *(const float2*)(src + 2 * i);
    dst[i] = pack2_bf16(v.x, v.y);
  }
}

__global__ void __launch_bounds__(NTHREADS)
compress_wmma_kernel(const float*          __restrict__ buffer,   // [BUF, H, D]
                     const unsigned short* __restrict__ gwb,      // [32, 4096] bf16
                     const int*            __restrict__ win_idx,  // [W, 32]
                     const int*            __restrict__ dest_idx, // [W]
                     float*                __restrict__ out,      // [CBUF, H, D]
                     int W) {
  extern __shared__ char smem[];
  unsigned short* Abf  = (unsigned short*)smem;                    // [MTILE][RSTRIDE]
  float*         cpart = (float*)(smem + LDS_ABF_BYTES);           // [4][2][8][32]
  float*         gl    = cpart + LDS_CPART_FLTS;                   // [MTILE][KWIN]

  const int h    = blockIdx.y;
  const int w0   = blockIdx.x * MTILE;
  const int tid  = threadIdx.x;
  const int lane = tid & 31;
  const int wv   = tid >> 5;

  // ------------------------------------------------------------------
  // Phase 1: gather 16 windows x 32 tokens x 128 f32, convert -> bf16 LDS
  // ------------------------------------------------------------------
  for (int row = tid; row < MTILE * KWIN; row += NTHREADS) {
    const int m = row >> 5;            // window within tile
    const int k = row & 31;            // token within window
    const int w = w0 + m;
    unsigned* dstu = (unsigned*)&Abf[m * RSTRIDE + k * HEAD_DIM];
    if (w < W) {
      const long tok = (long)win_idx[(long)w * KWIN + k];
      const float4* src = (const float4*)(buffer +
          tok * (long)(NUM_HEADS * HEAD_DIM) + (long)h * HEAD_DIM);
#pragma unroll 4
      for (int q = 0; q < HEAD_DIM / 4; ++q) {
        const float4 v = src[q];
        dstu[q * 2 + 0] = pack2_bf16(v.x, v.y);
        dstu[q * 2 + 1] = pack2_bf16(v.z, v.w);
      }
    } else {
#pragma unroll 4
      for (int q = 0; q < HEAD_DIM / 2; ++q) dstu[q] = 0u;
    }
  }
  __syncthreads();

  // ------------------------------------------------------------------
  // Phase 2: gate GEMM, M=16 windows, N=32 gate rows, K=4096 features.
  // Each wave owns K-chunks kk = wv*32, step 128. Two WMMAs per chunk.
  // Fragment builds are pure b128 loads:
  //   A (ISA 16-bit A layout): lane<16 holds K pairs {0..7,16..23},
  //     lane>=16 holds {8..15,24..31}  ->  two uint4 from LDS.
  //   B (same K order as A): lanes 0-15 hold K {0..7,16..23}, lanes 16-31 hold
  //     K {8..15,24..31}, pairs per VGPR ->  two uint4 from global (bf16 Gw).
  // ------------------------------------------------------------------
  const int am  = lane & 15;                 // A row (window)
  const int ahi = (lane >= 16) ? 8  : 0;     // A K-half select
  const int bhi = (lane >= 16) ? 8  : 0;     // B K-half select
  const int bn  = lane & 15;                 // B column (gate row)

  v8f c0 = {};
  v8f c1 = {};
  const unsigned short* arow = &Abf[am * RSTRIDE];
  const unsigned short* gw0  = gwb + (long)bn * FEAT + bhi;
  const unsigned short* gw1  = gwb + (long)(bn + 16) * FEAT + bhi;

#pragma unroll 2
  for (int kk = wv * 32; kk < FEAT; kk += 4 * 32) {
    union { v16bf v; uint4 q[2]; } a, b0, b1;
    a.q[0]  = *(const uint4*)(arow + kk + ahi);        // K pairs 0..7 (+ahi)
    a.q[1]  = *(const uint4*)(arow + kk + 16 + ahi);   // K pairs 16..23 (+ahi)
    b0.q[0] = *(const uint4*)(gw0 + kk);
    b0.q[1] = *(const uint4*)(gw0 + kk + 16);
    b1.q[0] = *(const uint4*)(gw1 + kk);
    b1.q[1] = *(const uint4*)(gw1 + kk + 16);
    c0 = __builtin_amdgcn_wmma_f32_16x16x32_bf16(false, a.v, false, b0.v,
                                                 (short)0, c0, false, false);
    c1 = __builtin_amdgcn_wmma_f32_16x16x32_bf16(false, a.v, false, b1.v,
                                                 (short)0, c1, false, false);
  }

  // Stash partial C per wave: cpart[wv][tile][r][lane]
#pragma unroll
  for (int r = 0; r < 8; ++r) {
    cpart[wv * 512 +       r * 32 + lane] = c0[r];
    cpart[wv * 512 + 256 + r * 32 + lane] = c1[r];
  }
  __syncthreads();

  // ------------------------------------------------------------------
  // Phase 3: cross-wave C reduction -> sigmoid gates in LDS [m][n]
  // C/D layout: VGPR r, lanes 0-15: (M=r, N=l); lanes 16-31: (M=r+8, N=l-16)
  // sigmoid via v_exp + v_rcp (gate activation; fast-math is appropriate)
  // ------------------------------------------------------------------
  for (int e = tid; e < 512; e += NTHREADS) {
    float s = cpart[e] + cpart[512 + e] + cpart[1024 + e] + cpart[1536 + e];
    s *= 0.015625f;                              // (32*128)^-0.5 = 1/64
    const float g = __builtin_amdgcn_rcpf(1.0f + __expf(-s));
    const int tile = e >> 8;
    const int r    = (e >> 5) & 7;
    const int l    = e & 31;
    const int mm   = r + ((l >= 16) ? 8 : 0);
    const int nn   = (l & 15) + tile * 16;
    gl[mm * KWIN + nn] = g;
  }
  __syncthreads();

  // ------------------------------------------------------------------
  // Phase 4: gated window reduction from staged LDS tile + scatter
  // ------------------------------------------------------------------
  for (int o = tid; o < MTILE * HEAD_DIM; o += NTHREADS) {
    const int mm = o >> 7;            // window within tile
    const int d  = o & 127;
    const int w  = w0 + mm;
    if (w >= W) continue;
    const unsigned short* trow = &Abf[mm * RSTRIDE + d];
    const float* grow = &gl[mm * KWIN];
    float acc = 0.0f;
#pragma unroll 8
    for (int k = 0; k < KWIN; ++k)
      acc += grow[k] * bf16_to_f32(trow[k * HEAD_DIM]);
    const long drow = (long)dest_idx[w];
    out[drow * (long)(NUM_HEADS * HEAD_DIM) + (long)h * HEAD_DIM + d] = acc;
  }
}

extern "C" void kernel_launch(void* const* d_in, const int* in_sizes, int n_in,
                              void* d_out, int out_size, void* d_ws, size_t ws_size,
                              hipStream_t stream) {
  const float* buffer     = (const float*)d_in[0];   // [BUF, H, D]
  const float* comp_init  = (const float*)d_in[1];   // [CBUF, H, D] (zeros)
  const float* gate_w     = (const float*)d_in[2];   // [32, 4096]
  const int*   win_idx    = (const int*)d_in[3];     // [W, 32]
  const int*   dest_idx   = (const int*)d_in[4];     // [W]
  float*       out        = (float*)d_out;

  const int W = in_sizes[4];

  // d_out is poisoned: initialize it from the (zero) compressed buffer input.
  {
    const long n = (long)out_size;
    long b = (n + NTHREADS - 1) / NTHREADS;
    const int blocks = b > 4096 ? 4096 : (int)b;
    compress_copy_init<<<blocks, NTHREADS, 0, stream>>>(comp_init, out, n);
  }

  // One-time gate_weight f32 -> bf16 into workspace (32*4096 = 131072 elems).
  unsigned* gwb = (unsigned*)d_ws;
  {
    const int npairs = (KWIN * FEAT) / 2;            // 65536 packed pairs
    gatew_to_bf16<<<(npairs + NTHREADS - 1) / NTHREADS, NTHREADS, 0, stream>>>(
        gate_w, gwb, npairs);
  }

  dim3 grid((W + MTILE - 1) / MTILE, NUM_HEADS, 1);
  compress_wmma_kernel<<<grid, NTHREADS, LDS_TOTAL_BYTES, stream>>>(
      buffer, (const unsigned short*)gwb, win_idx, dest_idx, out, W);
}
